// DistributedQCompilerGNN_75814762709515
// MI455X (gfx1250) — matrix operand order repaired and checked by reading, not compiled, not run on hardware
//
#include <hip/hip_runtime.h>
#include <hip/hip_bf16.h>
#include <stdint.h>

typedef _Float16 h16;
typedef __attribute__((ext_vector_type(16))) _Float16 v16h;
typedef __attribute__((ext_vector_type(8)))  float    v8f;

// ---------------------------------------------------------------- utility

__global__ void fill_kernel(float* __restrict__ p, float v, int n) {
    int i = blockIdx.x * blockDim.x + threadIdx.x;
    if (i < n) p[i] = v;
}

// ---------------------------------------------------------------- encoders
// out = relu(x[N,F] @ W[F,128] + b), also emit f16 copy for WMMA use.
__global__ void encode_kernel(const float* __restrict__ x, const float* __restrict__ W,
                              const float* __restrict__ b, float* __restrict__ out32,
                              h16* __restrict__ out16, int N, int F) {
    int idx = blockIdx.x * blockDim.x + threadIdx.x;
    if (idx >= N * 128) return;
    int n = idx >> 7, h = idx & 127;
    float acc = b[h];
    for (int f = 0; f < F; ++f) acc += x[n * F + f] * W[f * 128 + h];
    acc = fmaxf(acc, 0.f);
    out32[idx] = acc;
    if (out16) out16[idx] = (h16)acc;
}

// ---------------------------------------------------------------- seg-mean
__global__ void count_kernel(const int* __restrict__ dst, float* __restrict__ cnt, int E) {
    int e = blockIdx.x * blockDim.x + threadIdx.x;
    if (e < E) atomicAdd(&cnt[dst[e]], 1.f);
}

// 32 lanes per edge, float4 gather + 4 atomic adds.
__global__ void scatter_add_kernel(const float* __restrict__ feat, const int* __restrict__ src,
                                   const int* __restrict__ dst, float* __restrict__ accum, int E) {
    int idx = blockIdx.x * blockDim.x + threadIdx.x;
    if (idx >= E * 32) return;
    int e = idx >> 5;
    int h4 = (idx & 31) * 4;
    int s = src[e], d = dst[e];
    const float4 v = *reinterpret_cast<const float4*>(feat + (size_t)s * 128 + h4);
    float* p = accum + (size_t)d * 128 + h4;
    atomicAdd(p + 0, v.x);
    atomicAdd(p + 1, v.y);
    atomicAdd(p + 2, v.z);
    atomicAdd(p + 3, v.w);
}

// agg_f16 = accum / max(cnt,1)
__global__ void finalize_mean_kernel(const float* __restrict__ accum, const float* __restrict__ cnt,
                                     h16* __restrict__ agg, int N) {
    int idx = blockIdx.x * blockDim.x + threadIdx.x;
    if (idx >= N * 32) return;
    int n = idx >> 5;
    int h4 = (idx & 31) * 4;
    float s = 1.f / fmaxf(cnt[n], 1.f);
    const float4 v = *reinterpret_cast<const float4*>(accum + (size_t)n * 128 + h4);
    union { h16 h[4]; uint2 u; } pk;
    pk.h[0] = (h16)(v.x * s);
    pk.h[1] = (h16)(v.y * s);
    pk.h[2] = (h16)(v.z * s);
    pk.h[3] = (h16)(v.w * s);
    *reinterpret_cast<uint2*>(agg + (size_t)n * 128 + h4) = pk.u;
}

// ---------------------------------------------------------------- weight packing
// Pack (sum of up to 3) 128x128 f32 row-major K-by-N matrices into the WMMA
// B-fragment layout used here (contiguous k, same order as the A loads) for v_wmma_f32_16x16x32_f16:
//   tile (kt,nt) -> 32 lanes x 16 halfs; lane L covers K = kt*32 + 16*(L>>4) + j,
//   N = nt*16 + (L&15). Each lane's 16 halfs are contiguous in the packed buffer,
//   so the GEMM loads one B fragment with two coalesced b128 loads.
__global__ void pack_wfrag_kernel(const float* __restrict__ W0, const float* __restrict__ W1,
                                  const float* __restrict__ W2, int nT, h16* __restrict__ frag) {
    int idx = blockIdx.x * blockDim.x + threadIdx.x;
    if (idx >= 32 * 512) return;               // 4 K-tiles * 8 N-tiles * 32 lanes * 16 halfs
    int tile = idx >> 9;
    int rem  = idx & 511;
    int lane = rem >> 4;
    int j    = rem & 15;
    int kt = tile >> 3, nt = tile & 7;
    int k = kt * 32 + ((lane >> 4) << 4) + j;
    int n = nt * 16 + (lane & 15);
    float v = W0[k * 128 + n];
    if (nT >= 2) v += W1[k * 128 + n];
    if (nT >= 3) v += W2[k * 128 + n];
    frag[idx] = (h16)v;
}

__global__ void addvec_kernel(const float* __restrict__ a, const float* __restrict__ b,
                              const float* __restrict__ c, int nT, float* __restrict__ out, int n) {
    int i = blockIdx.x * blockDim.x + threadIdx.x;
    if (i >= n) return;
    float v = a[i] + b[i];
    if (nT >= 3) v += c[i];
    out[i] = v;
}

// ---------------------------------------------------------------- fused SAGE GEMM
// out = relu( bias + sum_t A_t[N,128] @ W_t[128,128] ), A/W in f16, f32 accumulate.
// Block = 256 threads = 8 waves; wave w owns column tile w, block owns one row tile.
// Per tile: K-loop of 4 x v_wmma_f32_16x16x32_f16 per term.
__global__ __launch_bounds__(256)
void sage_gemm_kernel(const h16* __restrict__ A0, const h16* __restrict__ A1,
                      const h16* __restrict__ A2, const h16* __restrict__ A3,
                      const h16* __restrict__ F0, const h16* __restrict__ F1,
                      const h16* __restrict__ F2, const h16* __restrict__ F3,
                      const float* __restrict__ bias, int nTerms,
                      float* __restrict__ out32, h16* __restrict__ out16) {
    const int lane = threadIdx.x & 31;
    const int nt   = threadIdx.x >> 5;     // column tile (0..7) -> N range nt*16..+15
    const int mt   = blockIdx.x;           // row tile
    const int lhi  = lane >> 4;            // lane group 0/1
    const int llo  = lane & 15;
    const int col  = nt * 16 + llo;

    v8f acc;
    {
        float bv = bias[col];
        #pragma unroll
        for (int r = 0; r < 8; ++r) acc[r] = bv;
    }

    const size_t rowOff = ((size_t)mt * 16 + llo) * 128;   // A row for this lane

    #pragma unroll
    for (int t = 0; t < 4; ++t) {
        if (t >= nTerms) break;                            // uniform scalar branch
        const h16* A = (t == 0) ? A0 : (t == 1) ? A1 : (t == 2) ? A2 : A3;
        const h16* F = (t == 0) ? F0 : (t == 1) ? F1 : (t == 2) ? F2 : F3;
        #pragma unroll
        for (int kt = 0; kt < 4; ++kt) {
            // A fragment (16x32 f16): lane holds row (llo), K contiguous at
            // kt*32 + 16*lhi .. +15  -> two b128 loads.
            union { v16h v; uint4 q[2]; } af, bf;
            const uint4* ap = reinterpret_cast<const uint4*>(A + rowOff + kt * 32 + lhi * 16);
            af.q[0] = ap[0];
            af.q[1] = ap[1];   // +8 halfs = +16 bytes
            // B fragment: pre-packed contiguous 16 halfs per lane.
            const uint4* bp = reinterpret_cast<const uint4*>(
                F + (((size_t)(kt * 8 + nt)) * 32 + lane) * 16);
            bf.q[0] = bp[0];
            bf.q[1] = bp[1];
            acc = __builtin_amdgcn_wmma_f32_16x16x32_f16(false, af.v, false, bf.v,
                                                         (short)0, acc, false, false);
        }
    }

    // D layout: vgpr r, lanes 0-15 -> M=r, lanes 16-31 -> M=r+8; N=llo.
    const size_t rowBase = (size_t)mt * 16;
    #pragma unroll
    for (int r = 0; r < 8; ++r) {
        float v = acc[r];
        v = v > 0.f ? v : 0.f;
        size_t o = (rowBase + r + lhi * 8) * 128 + col;
        out32[o] = v;
        out16[o] = (h16)v;
    }
}

// ---------------------------------------------------------------- attentional pooling
__global__ __launch_bounds__(256)
void gate_score_kernel(const float* __restrict__ hp, const float* __restrict__ Wg,
                       const float* __restrict__ bg, float* __restrict__ score, int N) {
    int wid  = (blockIdx.x * blockDim.x + threadIdx.x) >> 5;
    int lane = threadIdx.x & 31;
    if (wid >= N) return;
    float acc = 0.f;
    for (int h = lane; h < 128; h += 32) acc += hp[(size_t)wid * 128 + h] * Wg[h];
    for (int off = 16; off > 0; off >>= 1) acc += __shfl_down(acc, off, 32);
    if (lane == 0) score[wid] = acc + bg[0];
}

__device__ __forceinline__ unsigned encF(float f) {
    unsigned u = __float_as_uint(f);
    return (u & 0x80000000u) ? ~u : (u | 0x80000000u);
}
__device__ __forceinline__ float decF(unsigned u) {
    unsigned v = (u & 0x80000000u) ? (u & 0x7FFFFFFFu) : ~u;
    return __uint_as_float(v);
}

__global__ void segmax_kernel(const float* __restrict__ score, const int* __restrict__ batch,
                              unsigned* __restrict__ mEnc, int N) {
    int i = blockIdx.x * blockDim.x + threadIdx.x;
    if (i < N) atomicMax(&mEnc[batch[i]], encF(score[i]));
}

__global__ void expsum_kernel(const float* __restrict__ score, const int* __restrict__ batch,
                              const unsigned* __restrict__ mEnc, float* __restrict__ e,
                              float* __restrict__ denom, int N) {
    int i = blockIdx.x * blockDim.x + threadIdx.x;
    if (i >= N) return;
    int b = batch[i];
    float v = expf(score[i] - decF(mEnc[b]));
    e[i] = v;
    atomicAdd(&denom[b], v);
}

__global__ void emb_kernel(const float* __restrict__ hp, const float* __restrict__ e,
                           const float* __restrict__ denom, const int* __restrict__ batch,
                           float* __restrict__ emb, int N) {
    int idx = blockIdx.x * blockDim.x + threadIdx.x;
    if (idx >= N * 32) return;
    int n = idx >> 5;
    int h4 = (idx & 31) * 4;
    int b = batch[n];
    float alpha = e[n] / denom[b];
    const float4 v = *reinterpret_cast<const float4*>(hp + (size_t)n * 128 + h4);
    float* p = emb + (size_t)b * 128 + h4;
    atomicAdd(p + 0, alpha * v.x);
    atomicAdd(p + 1, alpha * v.y);
    atomicAdd(p + 2, alpha * v.z);
    atomicAdd(p + 3, alpha * v.w);
}

// ---------------------------------------------------------------- small dense heads (f32)
__global__ void dense_kernel(const float* __restrict__ X, const float* __restrict__ W,
                             const float* __restrict__ b, float* __restrict__ out,
                             int Brows, int K, int M, int doRelu) {
    int idx = blockIdx.x * blockDim.x + threadIdx.x;
    if (idx >= Brows * M) return;
    int r = idx / M, m = idx % M;
    float acc = b[m];
    for (int k = 0; k < K; ++k) acc += X[r * K + k] * W[k * M + m];
    if (doRelu) acc = fmaxf(acc, 0.f);
    out[idx] = acc;
}

// ---------------------------------------------------------------- host orchestration
extern "C" void kernel_launch(void* const* d_in, const int* in_sizes, int n_in,
                              void* d_out, int out_size, void* d_ws, size_t ws_size,
                              hipStream_t stream) {
    (void)in_sizes; (void)n_in; (void)out_size; (void)ws_size;
    constexpr int NP = 65536, NL = 32768, NG = 131072, H = 128, B = 128, A = 256;
    constexpr int EIN = 1048576, EIT = 262144, EMP = 65536, EPT = 262144, EDP = 524288;

    const float* x_phys = (const float*)d_in[0];
    const float* x_log  = (const float*)d_in[1];
    const float* x_gate = (const float*)d_in[2];
    const int*   batch  = (const int*)d_in[3];
    const int* eiS[5] = {(const int*)d_in[4], (const int*)d_in[6], (const int*)d_in[8],
                         (const int*)d_in[10], (const int*)d_in[12]};
    const int* eiD[5] = {(const int*)d_in[5], (const int*)d_in[7], (const int*)d_in[9],
                         (const int*)d_in[11], (const int*)d_in[13]};
    const float* W_phys = (const float*)d_in[14]; const float* b_phys = (const float*)d_in[15];
    const float* W_log  = (const float*)d_in[16]; const float* b_log  = (const float*)d_in[17];
    const float* W_gate = (const float*)d_in[18]; const float* b_gate = (const float*)d_in[19];
    const float* Wl[2] = {(const float*)d_in[20], (const float*)d_in[23]};
    const float* bl[2] = {(const float*)d_in[21], (const float*)d_in[24]};
    const float* Wr[2] = {(const float*)d_in[22], (const float*)d_in[25]};
    const float* Wg  = (const float*)d_in[26]; const float* bg  = (const float*)d_in[27];
    const float* Wa1 = (const float*)d_in[28]; const float* ba1 = (const float*)d_in[29];
    const float* Wa2 = (const float*)d_in[30]; const float* ba2 = (const float*)d_in[31];
    const float* Wc1 = (const float*)d_in[32]; const float* bc1 = (const float*)d_in[33];
    const float* Wc2 = (const float*)d_in[34]; const float* bc2 = (const float*)d_in[35];
    float* out = (float*)d_out;

    // workspace bump allocator
    char* base = (char*)d_ws;
    size_t off = 0;
    auto alloc = [&](size_t bytes) -> char* {
        off = (off + 255) & ~(size_t)255;
        char* p = base + off;
        off += bytes;
        return p;
    };

    float* hp32a = (float*)alloc((size_t)NP * H * 4); h16* hp16a = (h16*)alloc((size_t)NP * H * 2);
    float* hp32b = (float*)alloc((size_t)NP * H * 4); h16* hp16b = (h16*)alloc((size_t)NP * H * 2);
    float* hg32a = (float*)alloc((size_t)NG * H * 4); h16* hg16a = (h16*)alloc((size_t)NG * H * 2);
    float* hg32b = (float*)alloc((size_t)NG * H * 4); h16* hg16b = (h16*)alloc((size_t)NG * H * 2);
    float* hl32  = (float*)alloc((size_t)NL * H * 4);
    h16* aggP[3]; for (int i = 0; i < 3; ++i) aggP[i] = (h16*)alloc((size_t)NP * H * 2);
    h16* aggG[2]; for (int i = 0; i < 2; ++i) aggG[i] = (h16*)alloc((size_t)NG * H * 2);
    float* accum = (float*)alloc((size_t)NG * H * 4);
    float* cnt[5];
    cnt[0] = (float*)alloc(NP * 4); cnt[1] = (float*)alloc(NP * 4); cnt[2] = (float*)alloc(NP * 4);
    cnt[3] = (float*)alloc(NG * 4); cnt[4] = (float*)alloc(NG * 4);
    h16* wfrag[2][7];
    for (int L = 0; L < 2; ++L) for (int t = 0; t < 7; ++t) wfrag[L][t] = (h16*)alloc(16384 * 2);
    float* biasP[2]; float* biasG[2];
    for (int L = 0; L < 2; ++L) { biasP[L] = (float*)alloc(H * 4); biasG[L] = (float*)alloc(H * 4); }
    float* score = (float*)alloc((size_t)NP * 4);
    float* eexp  = (float*)alloc((size_t)NP * 4);
    unsigned* mEnc = (unsigned*)alloc(B * 4);
    float* denom = (float*)alloc(B * 4);
    float* emb = (float*)alloc((size_t)B * H * 4);
    float* ha  = (float*)alloc((size_t)B * H * 4);
    float* hc  = (float*)alloc((size_t)B * H * 4);

    const int Edges[5] = {EIN, EIT, EMP, EPT, EDP};
    const int Ndst[5]  = {NP, NP, NP, NG, NG};

    auto cdiv = [](long long a, long long b) { return (int)((a + b - 1) / b); };

    // edge counts (layer-invariant)
    for (int t = 0; t < 5; ++t) {
        fill_kernel<<<cdiv(Ndst[t], 256), 256, 0, stream>>>(cnt[t], 0.f, Ndst[t]);
        count_kernel<<<cdiv(Edges[t], 256), 256, 0, stream>>>(eiD[t], cnt[t], Edges[t]);
    }

    // encoders
    encode_kernel<<<cdiv((long long)NP * H, 256), 256, 0, stream>>>(x_phys, W_phys, b_phys, hp32a, hp16a, NP, 5);
    encode_kernel<<<cdiv((long long)NL * H, 256), 256, 0, stream>>>(x_log, W_log, b_log, hl32, (h16*)nullptr, NL, 1);
    encode_kernel<<<cdiv((long long)NG * H, 256), 256, 0, stream>>>(x_gate, W_gate, b_gate, hg32a, hg16a, NG, 2);

    // pack weights: 5 Wl per layer; combined Wr (phys = Wr0+Wr1+Wr2, gate = Wr3+Wr4); bias sums
    for (int L = 0; L < 2; ++L) {
        for (int t = 0; t < 5; ++t)
            pack_wfrag_kernel<<<64, 256, 0, stream>>>(Wl[L] + (size_t)t * H * H, nullptr, nullptr, 1, wfrag[L][t]);
        pack_wfrag_kernel<<<64, 256, 0, stream>>>(Wr[L], Wr[L] + (size_t)H * H, Wr[L] + (size_t)2 * H * H, 3, wfrag[L][5]);
        pack_wfrag_kernel<<<64, 256, 0, stream>>>(Wr[L] + (size_t)3 * H * H, Wr[L] + (size_t)4 * H * H, nullptr, 2, wfrag[L][6]);
        addvec_kernel<<<1, 128, 0, stream>>>(bl[L], bl[L] + H, bl[L] + 2 * H, 3, biasP[L], H);
        addvec_kernel<<<1, 128, 0, stream>>>(bl[L] + 3 * H, bl[L] + 4 * H, nullptr, 2, biasG[L], H);
    }

    float* inP32 = hp32a; h16* inP16 = hp16a; float* inG32 = hg32a; h16* inG16 = hg16a;
    float* outP32 = hp32b; h16* outP16 = hp16b; float* outG32 = hg32b; h16* outG16 = hg16b;

    for (int L = 0; L < 2; ++L) {
        const float* srcFeat[5] = {inP32, inP32, hl32, hl32, inG32};
        h16* aggDst[5] = {aggP[0], aggP[1], aggP[2], aggG[0], aggG[1]};
        for (int t = 0; t < 5; ++t) {
            int n = Ndst[t], E = Edges[t];
            fill_kernel<<<cdiv((long long)n * H, 256), 256, 0, stream>>>(accum, 0.f, n * H);
            scatter_add_kernel<<<cdiv((long long)E * 32, 256), 256, 0, stream>>>(srcFeat[t], eiS[t], eiD[t], accum, E);
            finalize_mean_kernel<<<cdiv((long long)n * 32, 256), 256, 0, stream>>>(accum, cnt[t], aggDst[t], n);
        }
        // phys: 4 fused GEMM terms; gate: 3 (4th slot is an unused dummy)
        sage_gemm_kernel<<<NP / 16, 256, 0, stream>>>(aggP[0], aggP[1], aggP[2], inP16,
            wfrag[L][0], wfrag[L][1], wfrag[L][2], wfrag[L][5], biasP[L], 4, outP32, outP16);
        sage_gemm_kernel<<<NG / 16, 256, 0, stream>>>(aggG[0], aggG[1], inG16, aggG[0],
            wfrag[L][3], wfrag[L][4], wfrag[L][6], wfrag[L][3], biasG[L], 3, outG32, outG16);
        { float* t = inP32; inP32 = outP32; outP32 = t; }
        { h16*   t = inP16; inP16 = outP16; outP16 = t; }
        { float* t = inG32; inG32 = outG32; outG32 = t; }
        { h16*   t = inG16; inG16 = outG16; outG16 = t; }
    }

    // attentional pooling over physical nodes (inP32 now holds layer-2 output)
    gate_score_kernel<<<cdiv((long long)NP * 32, 256), 256, 0, stream>>>(inP32, Wg, bg, score, NP);
    fill_kernel<<<1, 256, 0, stream>>>((float*)mEnc, 0.f, B);   // bits 0 == encoded -inf
    fill_kernel<<<1, 256, 0, stream>>>(denom, 0.f, B);
    fill_kernel<<<cdiv((long long)B * H, 256), 256, 0, stream>>>(emb, 0.f, B * H);
    segmax_kernel<<<cdiv(NP, 256), 256, 0, stream>>>(score, batch, mEnc, NP);
    expsum_kernel<<<cdiv(NP, 256), 256, 0, stream>>>(score, batch, mEnc, eexp, denom, NP);
    emb_kernel<<<cdiv((long long)NP * 32, 256), 256, 0, stream>>>(inP32, eexp, denom, batch, emb, NP);

    // actor / critic heads
    dense_kernel<<<cdiv((long long)B * H, 256), 256, 0, stream>>>(emb, Wa1, ba1, ha, B, H, H, 1);
    dense_kernel<<<cdiv((long long)B * A, 256), 256, 0, stream>>>(ha, Wa2, ba2, out, B, H, A, 0);
    dense_kernel<<<cdiv((long long)B * H, 256), 256, 0, stream>>>(emb, Wc1, bc1, hc, B, H, H, 1);
    dense_kernel<<<cdiv(B, 256), 256, 0, stream>>>(hc, Wc2, bc2, out + (size_t)B * A, B, H, 1, 0);
}
